// NeighConv1_70944269796241
// MI455X (gfx1250) — matrix operand order repaired and checked by reading, not compiled, not run on hardware
//
#include <hip/hip_runtime.h>
#include <hip/hip_bf16.h>

// CDNA5 (gfx1250) types for WMMA
typedef __attribute__((ext_vector_type(16))) __bf16 v16bf;
typedef __attribute__((ext_vector_type(8)))  float  v8f;

#define EMB     256
#define BDIM    4096
#define MAXLEN  50
#define NLAYERS 3
#define BK      32
#define LDT     40   // padded LDS row stride in bf16 elements (80B = 20 banks, 16B aligned)

// ---------- bf16 split helpers ----------
__device__ __forceinline__ unsigned short bf16_rn(float x) {
  unsigned int u = __float_as_uint(x);
  return (unsigned short)((u + 0x7FFFu + ((u >> 16) & 1u)) >> 16);
}
__device__ __forceinline__ float bf16_up(unsigned short h) {
  return __uint_as_float(((unsigned int)h) << 16);
}
__device__ __forceinline__ void split2(float x, unsigned short& hi, unsigned short& lo) {
  hi = bf16_rn(x);
  lo = bf16_rn(x - bf16_up(hi));
}
__device__ __forceinline__ v16bf ldfrag2(const unsigned short* p0, const unsigned short* p1) {
  union { uint4 q[2]; v16bf v; } u;
  u.q[0] = *(const uint4*)p0;   // 8 bf16, 16B, ds_load_b128
  u.q[1] = *(const uint4*)p1;
  return u.v;
}

// ---------- kernel 1: ragged gather + masked mean ----------
__global__ __launch_bounds__(256)
void sess_mean_kernel(const float* __restrict__ emb, const int* __restrict__ items,
                      const int* __restrict__ slen, float* __restrict__ cur,
                      float* __restrict__ acc) {
  const int b = blockIdx.x;
  const int e = threadIdx.x;          // one thread per embedding dim (EMB == 256)
  int len = slen[b];
  if (len < 1) len = 1;
  if (len > MAXLEN) len = MAXLEN;
  float s = 0.f;
  for (int t = 0; t < len; ++t) {
    int it = items[b * MAXLEN + t];
    s += emb[(size_t)it * EMB + e];
  }
  float v = s / (float)len;
  cur[(size_t)b * EMB + e] = v;
  acc[(size_t)b * EMB + e] = v;
}

// ---------- kernel 2: C = A(f32,MxK) @ B(f32,KxN) via split-bf16 WMMA ----------
// Tile 128x128x32, 8 waves (2x4), 64x32 per wave => 4x2 WMMA tiles, 3 products each.
__global__ __launch_bounds__(256)
void gemm_bf16x3_kernel(const float* __restrict__ Ag, const float* __restrict__ Bg,
                        float* __restrict__ Cg, float* __restrict__ accOut,
                        int M, int N, int K) {
  __shared__ __align__(16) unsigned short sAhi[128 * LDT];
  __shared__ __align__(16) unsigned short sAlo[128 * LDT];
  __shared__ __align__(16) unsigned short sBhi[128 * LDT];  // B stored transposed: [n][k]
  __shared__ __align__(16) unsigned short sBlo[128 * LDT];

  const int t    = threadIdx.x;
  const int lane = t & 31;
  const int w    = t >> 5;
  const int wm   = w >> 2;        // 0..1  (64-row band)
  const int wn   = w & 3;         // 0..3  (32-col band)
  const int half = lane >> 4;     // wave32 half
  const int l16  = lane & 15;

  const int m0 = blockIdx.y * 128;
  const int n0 = blockIdx.x * 128;

  // global->reg staging mappings
  const int ar  = t >> 3;         // A: base row 0..31 (per q add q*32)
  const int ac  = (t & 7) * 4;    // A: col (float4)
  const int bn  = t & 127;        // B: local column n
  const int bkq = t >> 7;         // B: 0..1 -> k sub-range of 16

  float4 aS[4];
  float  bS[16];

  auto loadA = [&](int k0) {
#pragma unroll
    for (int q = 0; q < 4; ++q)
      aS[q] = *(const float4*)&Ag[(size_t)(m0 + q * 32 + ar) * K + k0 + ac];
  };
  auto loadB = [&](int k0) {
#pragma unroll
    for (int i = 0; i < 16; ++i)
      bS[i] = Bg[(size_t)(k0 + bkq * 16 + i) * N + n0 + bn];
  };
  auto storeLDS = [&]() {
#pragma unroll
    for (int q = 0; q < 4; ++q) {
      unsigned short h[4], l[4];
      split2(aS[q].x, h[0], l[0]);
      split2(aS[q].y, h[1], l[1]);
      split2(aS[q].z, h[2], l[2]);
      split2(aS[q].w, h[3], l[3]);
      const int off = (q * 32 + ar) * LDT + ac;
      uint2 ph, pl;
      ph.x = (unsigned)h[0] | ((unsigned)h[1] << 16);
      ph.y = (unsigned)h[2] | ((unsigned)h[3] << 16);
      pl.x = (unsigned)l[0] | ((unsigned)l[1] << 16);
      pl.y = (unsigned)l[2] | ((unsigned)l[3] << 16);
      *(uint2*)&sAhi[off] = ph;
      *(uint2*)&sAlo[off] = pl;
    }
    {
      unsigned short h[16], l[16];
#pragma unroll
      for (int i = 0; i < 16; ++i) split2(bS[i], h[i], l[i]);
      unsigned int ph[8], pl[8];
#pragma unroll
      for (int i = 0; i < 8; ++i) {
        ph[i] = (unsigned)h[2 * i] | ((unsigned)h[2 * i + 1] << 16);
        pl[i] = (unsigned)l[2 * i] | ((unsigned)l[2 * i + 1] << 16);
      }
      const int off = bn * LDT + bkq * 16;  // transposed: row n, 16 consecutive k
      *(uint4*)&sBhi[off]     = make_uint4(ph[0], ph[1], ph[2], ph[3]);
      *(uint4*)&sBhi[off + 8] = make_uint4(ph[4], ph[5], ph[6], ph[7]);
      *(uint4*)&sBlo[off]     = make_uint4(pl[0], pl[1], pl[2], pl[3]);
      *(uint4*)&sBlo[off + 8] = make_uint4(pl[4], pl[5], pl[6], pl[7]);
    }
  };

  v8f accv[4][2];
#pragma unroll
  for (int mt = 0; mt < 4; ++mt)
#pragma unroll
    for (int nt = 0; nt < 2; ++nt)
      accv[mt][nt] = (v8f){0.f, 0.f, 0.f, 0.f, 0.f, 0.f, 0.f, 0.f};

  const int KT = K / BK;
  loadA(0);
  loadB(0);
  storeLDS();
  __syncthreads();

  for (int kt = 0; kt < KT; ++kt) {
    if (kt + 1 < KT) {                       // overlap next-tile global loads with WMMA
      loadA((kt + 1) * BK);
      loadB((kt + 1) * BK);
    }

    // B fragments: lane(h,n): j<8 -> K=8h+j ; j>=8 -> K=16+8h+(j-8)
    v16bf bh[2], bl[2];
#pragma unroll
    for (int nt = 0; nt < 2; ++nt) {
      const int off = (wn * 32 + nt * 16 + l16) * LDT + half * 8;
      bh[nt] = ldfrag2(&sBhi[off], &sBhi[off + 16]);
      bl[nt] = ldfrag2(&sBlo[off], &sBlo[off + 16]);
    }
    // A fragments: lane(h,m): j<8 -> K=8h+j ; j>=8 -> K=16+8h+(j-8)
#pragma unroll
    for (int mt = 0; mt < 4; ++mt) {
      const int arow = (wm * 64 + mt * 16 + l16) * LDT;
      v16bf ah = ldfrag2(&sAhi[arow + half * 8], &sAhi[arow + 16 + half * 8]);
      v16bf al = ldfrag2(&sAlo[arow + half * 8], &sAlo[arow + 16 + half * 8]);
#pragma unroll
      for (int nt = 0; nt < 2; ++nt) {
        accv[mt][nt] = __builtin_amdgcn_wmma_f32_16x16x32_bf16(
            false, ah, false, bh[nt], (short)0, accv[mt][nt], false, false);
        accv[mt][nt] = __builtin_amdgcn_wmma_f32_16x16x32_bf16(
            false, ah, false, bl[nt], (short)0, accv[mt][nt], false, false);
        accv[mt][nt] = __builtin_amdgcn_wmma_f32_16x16x32_bf16(
            false, al, false, bh[nt], (short)0, accv[mt][nt], false, false);
      }
    }
    __syncthreads();
    if (kt + 1 < KT) {
      storeLDS();
      __syncthreads();
    }
  }

  // epilogue: C/D layout -> row = r + 8*half, col = lane&15 within each 16x16 tile
#pragma unroll
  for (int mt = 0; mt < 4; ++mt)
#pragma unroll
    for (int nt = 0; nt < 2; ++nt) {
      union { v8f v; float f[8]; } u;
      u.v = accv[mt][nt];
#pragma unroll
      for (int r = 0; r < 8; ++r) {
        const int row = m0 + wm * 64 + mt * 16 + half * 8 + r;
        const int col = n0 + wn * 32 + nt * 16 + l16;
        const size_t idx = (size_t)row * N + col;
        Cg[idx] = u.f[r];
        if (accOut != nullptr) accOut[idx] += u.f[r];
      }
    }
}

// ---------- kernel 3: out = acc / (LAYERS+1) ----------
__global__ __launch_bounds__(256)
void finalize_kernel(const float* __restrict__ acc, float* __restrict__ out, int n) {
  int i = blockIdx.x * blockDim.x + threadIdx.x;
  if (i < n) out[i] = acc[i] * 0.25f;
}

extern "C" void kernel_launch(void* const* d_in, const int* in_sizes, int n_in,
                              void* d_out, int out_size, void* d_ws, size_t ws_size,
                              hipStream_t stream) {
  const float* emb   = (const float*)d_in[0];   // [50000,256]
  const float* Dmat  = (const float*)d_in[1];   // [4096,4096]
  const float* Amat  = (const float*)d_in[2];   // [4096,4096]
  const int*   items = (const int*)d_in[3];     // [4096,50]
  const int*   slen  = (const int*)d_in[4];     // [4096]
  float* out = (float*)d_out;                   // [4096,256]

  const size_t SESS = (size_t)BDIM * EMB;       // 1M floats
  float* cur = (float*)d_ws;
  float* tmp = cur + SESS;
  float* acc = tmp + SESS;

  // 1) session mean -> cur, acc
  sess_mean_kernel<<<BDIM, 256, 0, stream>>>(emb, items, slen, cur, acc);

  // 2) LightGCN propagation, re-associated: cur = D @ (A @ cur); acc += cur
  dim3 grid(EMB / 128, BDIM / 128);             // (2, 32)
  for (int l = 0; l < NLAYERS; ++l) {
    gemm_bf16x3_kernel<<<grid, 256, 0, stream>>>(Amat, cur, tmp, nullptr, BDIM, EMB, BDIM);
    gemm_bf16x3_kernel<<<grid, 256, 0, stream>>>(Dmat, tmp, cur, acc, BDIM, EMB, BDIM);
  }

  // 3) out = acc / (LAYERS + 1)
  const int n = (int)SESS;
  finalize_kernel<<<(n + 255) / 256, 256, 0, stream>>>(acc, out, n);
}
